// KGEncoder_76647986365230
// MI455X (gfx1250) — compile-verified, matrix operand order checked by reading, not run on hardware
//
#include <hip/hip_runtime.h>
#include <hip/hip_bf16.h>

// ---------------------------------------------------------------------------
// Problem constants (from the reference)
// ---------------------------------------------------------------------------
constexpr int N_  = 20000;
constexpr int E_  = 320000;
constexpr int D_  = 256;
constexpr int De_ = 64;
constexpr int H_  = 8;
constexpr int L_  = 3;
constexpr int Q_  = 3;
constexpr int K_  = 512;
constexpr int FD_ = 2048;
constexpr int NG_ = 64;
constexpr int C_  = 32;              // D_/H_
constexpr float EPS_ = 1e-5f;

typedef __bf16 bf16_t;
typedef __attribute__((ext_vector_type(16))) __bf16 v16bf;
typedef __attribute__((ext_vector_type(8)))  __bf16 v8bf;
typedef __attribute__((ext_vector_type(8)))  float  v8f;

// ---------------------------------------------------------------------------
// Generic WMMA GEMM: C[M,Nc] = A[M,K] * B[K,Nc] (+bias) (optional ReLU)
// A,B fp32 row-major, converted to bf16 in LDS, fp32 accumulate via
// v_wmma_f32_16x16x32_bf16. Block tile 128x64, 8 waves (wave32), each wave
// owns a 32x32 output tile (2x2 WMMA accumulators).
// A tile stored [row][k], B tile stored TRANSPOSED [col][k]: both fragments
// are two aligned 8x bf16 contiguous runs per lane -> ds_load_b128 pairs.
// Requires: K % 32 == 0, Nc % 64 == 0 (true for all call sites).
// ---------------------------------------------------------------------------
#define BM 128
#define BN 64
#define BK 32
#define LDK 40   // padded k-stride (elements); 80B rows keep 16B alignment

// Build a 16-element bf16 fragment from two contiguous 8-element runs:
// elements 0..7  = rowbase[lh*8 .. lh*8+7]
// elements 8..15 = rowbase[16+lh*8 .. 16+lh*8+7]
__device__ __forceinline__ v16bf load_frag(const bf16_t* rowbase, int lh) {
    v8bf lo = *(const v8bf*)(rowbase + lh * 8);
    v8bf hi = *(const v8bf*)(rowbase + 16 + lh * 8);
    return __builtin_shufflevector(lo, hi, 0,1,2,3,4,5,6,7,8,9,10,11,12,13,14,15);
}

__global__ __launch_bounds__(256) void gemm_bf16_wmma(
    const float* __restrict__ A, const float* __restrict__ B,
    const float* __restrict__ bias, float* __restrict__ C,
    int M, int K, int Nc, int applyRelu)
{
    __shared__ bf16_t As[BM * LDK];   // [row][k]
    __shared__ bf16_t Bs[BN * LDK];   // transposed: [col][k]

    const int tid  = threadIdx.x;
    const int wave = tid >> 5;
    const int lane = tid & 31;
    const int wm   = wave & 3;   // row group of wave
    const int wn   = wave >> 2;  // col group of wave

    const int row0 = blockIdx.x * BM;
    const int col0 = blockIdx.y * BN;

    v8f zero = {0.f,0.f,0.f,0.f,0.f,0.f,0.f,0.f};
    v8f acc[2][2];
    acc[0][0]=zero; acc[0][1]=zero; acc[1][0]=zero; acc[1][1]=zero;

    const int lm = lane & 15;
    const int lh = lane >> 4;

    for (int k0 = 0; k0 < K; k0 += BK) {
        // --- stage A tile (128x32) as bf16, one v8bf (ds_store_b128) per seg
        #pragma unroll
        for (int it = 0; it < 2; ++it) {
            int seg = tid + it * 256;          // 512 segments of 8
            int r   = seg >> 2;                // row 0..127
            int c8  = (seg & 3) * 8;           // k offset 0/8/16/24
            int gr  = row0 + r;
            v8bf val;
            if (gr < M) {
                const float* ap = A + (size_t)gr * K + k0 + c8;
                #pragma unroll
                for (int j = 0; j < 8; ++j) val[j] = (bf16_t)ap[j];
                if (k0 + BK < K)
                    __builtin_prefetch(ap + BK, 0, 1);
            } else {
                #pragma unroll
                for (int j = 0; j < 8; ++j) val[j] = (bf16_t)0.f;
            }
            *(v8bf*)(&As[r * LDK + c8]) = val;
        }
        // --- stage B tile (32x64) TRANSPOSED as bf16: 256 segments of 8
        {
            int n  = tid >> 2;                 // col 0..63
            int k8 = (tid & 3) * 8;            // k offset 0/8/16/24
            const float* bp = B + (size_t)(k0 + k8) * Nc + col0 + n;
            v8bf val;
            #pragma unroll
            for (int j = 0; j < 8; ++j) val[j] = (bf16_t)bp[(size_t)j * Nc];
            *(v8bf*)(&Bs[n * LDK + k8]) = val;
        }
        __syncthreads();

        // --- fragments: 2 x ds_load_b128 each
        v16bf afrag[2], bfrag[2];
        #pragma unroll
        for (int mi = 0; mi < 2; ++mi)
            afrag[mi] = load_frag(&As[(wm * 32 + mi * 16 + lm) * LDK], lh);
        #pragma unroll
        for (int ni = 0; ni < 2; ++ni)
            bfrag[ni] = load_frag(&Bs[(wn * 32 + ni * 16 + lm) * LDK], lh);

        #pragma unroll
        for (int mi = 0; mi < 2; ++mi)
            #pragma unroll
            for (int ni = 0; ni < 2; ++ni)
                acc[mi][ni] = __builtin_amdgcn_wmma_f32_16x16x32_bf16(
                    false, afrag[mi], false, bfrag[ni],
                    (short)0, acc[mi][ni], false, false);
        __syncthreads();
    }

    // store: lane l -> n = l&15 ; VGPR v -> m = v + 8*(l>>4)
    #pragma unroll
    for (int mi = 0; mi < 2; ++mi) {
        #pragma unroll
        for (int ni = 0; ni < 2; ++ni) {
            int cc = col0 + wn * 32 + ni * 16 + lm;
            float bv = bias ? bias[cc] : 0.f;
            #pragma unroll
            for (int vv = 0; vv < 8; ++vv) {
                int rr = row0 + wm * 32 + mi * 16 + vv + 8 * lh;
                if (rr < M) {
                    float val = acc[mi][ni][vv] + bv;
                    if (applyRelu) val = val > 0.f ? val : 0.f;
                    C[(size_t)rr * Nc + cc] = val;
                }
            }
        }
    }
}

// ---------------------------------------------------------------------------
// utility kernels
// ---------------------------------------------------------------------------
__global__ void fill_f32(float* p, float v, long n) {
    long i = (long)blockIdx.x * blockDim.x + threadIdx.x;
    if (i < n) p[i] = v;
}
__global__ void fill_u32(unsigned* p, unsigned v, long n) {
    long i = (long)blockIdx.x * blockDim.x + threadIdx.x;
    if (i < n) p[i] = v;
}

// order-preserving float<->uint key for atomic max on floats
__device__ __forceinline__ unsigned fkey(float f) {
    unsigned b = __float_as_uint(f);
    return (b & 0x80000000u) ? ~b : (b | 0x80000000u);
}
__device__ __forceinline__ float fdec(unsigned k) {
    unsigned b = (k & 0x80000000u) ? (k ^ 0x80000000u) : ~k;
    return __uint_as_float(b);
}

// ---------------------------------------------------------------------------
// edge attention pipeline
// ---------------------------------------------------------------------------
__global__ void edge_logits(const int* __restrict__ src, const int* __restrict__ dst,
                            const float* __restrict__ q, const float* __restrict__ kb,
                            const float* __restrict__ eb,
                            float* __restrict__ att, unsigned* __restrict__ mkeys)
{
    long idx = (long)blockIdx.x * 256 + threadIdx.x;   // (edge, head)
    if (idx >= (long)E_ * H_) return;
    int eg = (int)(idx >> 3);
    int h  = (int)(idx & 7);
    int s = src[eg], d = dst[eg];
    const float4* qp = (const float4*)(q  + (size_t)d * D_ + h * C_);
    const float4* kp = (const float4*)(kb + (size_t)s * D_ + h * C_);
    const float4* ep = (const float4*)(eb + (size_t)eg * D_ + h * C_);
    float acc = 0.f;
    #pragma unroll
    for (int c = 0; c < C_ / 4; ++c) {
        float4 qv = qp[c], kv = kp[c], ev = ep[c];
        acc += qv.x * (kv.x + ev.x) + qv.y * (kv.y + ev.y)
             + qv.z * (kv.z + ev.z) + qv.w * (kv.w + ev.w);
    }
    acc *= 0.17677669529663687f;   // 1/sqrt(32)
    att[idx] = acc;
    atomicMax(&mkeys[d * H_ + h], fkey(acc));
}

__global__ void edge_softmax_num(const int* __restrict__ dst,
                                 float* __restrict__ att,
                                 const unsigned* __restrict__ mkeys,
                                 float* __restrict__ denom)
{
    long idx = (long)blockIdx.x * 256 + threadIdx.x;
    if (idx >= (long)E_ * H_) return;
    int eg = (int)(idx >> 3);
    int h  = (int)(idx & 7);
    int d = dst[eg];
    float m = fdec(mkeys[d * H_ + h]);
    float a = expf(att[idx] - m);
    att[idx] = a;
    atomicAdd(&denom[d * H_ + h], a);
}

__global__ void edge_message(const int* __restrict__ src, const int* __restrict__ dst,
                             const float* __restrict__ vbuf, const float* __restrict__ eb,
                             const float* __restrict__ att, const float* __restrict__ denom,
                             float* __restrict__ xo)
{
    long idx = (long)blockIdx.x * 256 + threadIdx.x;   // (edge, feature)
    if (idx >= (long)E_ * D_) return;
    int eg = (int)(idx >> 8);
    int dd = (int)(idx & 255);
    int h  = dd >> 5;
    int s = src[eg], d = dst[eg];
    float alpha = att[(size_t)eg * H_ + h] / denom[(size_t)d * H_ + h];
    float val = (vbuf[(size_t)s * D_ + dd] + eb[(size_t)eg * D_ + dd]) * alpha;
    atomicAdd(&xo[(size_t)d * D_ + dd], val);
}

// ---------------------------------------------------------------------------
// GraphNorm passes + counts + pooling
// ---------------------------------------------------------------------------
__global__ void count_kernel(const int* __restrict__ batch, float* __restrict__ cnt) {
    int n = blockIdx.x * 256 + threadIdx.x;
    if (n < N_) atomicAdd(&cnt[batch[n]], 1.f);
}

__global__ void gn_pass1(const float* __restrict__ xo, const float* __restrict__ xin,
                         const int* __restrict__ batch,
                         float* __restrict__ y, float* __restrict__ gsum)
{
    long idx = (long)blockIdx.x * 256 + threadIdx.x;
    if (idx >= (long)N_ * D_) return;
    int n = (int)(idx >> 8), d = (int)(idx & 255);
    float v = xo[idx] + xin[idx];
    y[idx] = v;
    atomicAdd(&gsum[batch[n] * D_ + d], v);
}

__global__ void gn_pass2(const float* __restrict__ y, const float* __restrict__ gsum,
                         const float* __restrict__ cnt, const float* __restrict__ ms,
                         const int* __restrict__ batch,
                         float* __restrict__ xc, float* __restrict__ vsum)
{
    long idx = (long)blockIdx.x * 256 + threadIdx.x;
    if (idx >= (long)N_ * D_) return;
    int n = (int)(idx >> 8), d = (int)(idx & 255);
    int g = batch[n];
    float c = fmaxf(cnt[g], 1.f);
    float mean = gsum[g * D_ + d] / c;
    float v = y[idx] - ms[d] * mean;
    xc[idx] = v;
    atomicAdd(&vsum[g * D_ + d], v * v);
}

__global__ void gn_pass3(const float* __restrict__ xc, const float* __restrict__ vsum,
                         const float* __restrict__ cnt, const float* __restrict__ w,
                         const float* __restrict__ b, const int* __restrict__ batch,
                         float* __restrict__ xout)
{
    long idx = (long)blockIdx.x * 256 + threadIdx.x;
    if (idx >= (long)N_ * D_) return;
    int n = (int)(idx >> 8), d = (int)(idx & 255);
    int g = batch[n];
    float c = fmaxf(cnt[g], 1.f);
    float var = vsum[g * D_ + d] / c;
    xout[idx] = xc[idx] * rsqrtf(var + EPS_) * w[d] + b[d];
}

__global__ void pool_accum(const float* __restrict__ x, const int* __restrict__ batch,
                           float* __restrict__ gsum)
{
    long idx = (long)blockIdx.x * 256 + threadIdx.x;
    if (idx >= (long)N_ * D_) return;
    int n = (int)(idx >> 8), d = (int)(idx & 255);
    atomicAdd(&gsum[batch[n] * D_ + d], x[idx]);
}

// ---------------------------------------------------------------------------
// Directional VQ: one block (256 threads, thread==feature) per graph
// ---------------------------------------------------------------------------
__global__ __launch_bounds__(256) void vq_kernel(
    const float* __restrict__ gsum, const float* __restrict__ cnt,
    const float* __restrict__ codebooks,
    float* __restrict__ tokens, float* __restrict__ out_idx, float* __restrict__ loss)
{
    const int g   = blockIdx.x;
    const int tid = threadIdx.x;       // == feature index (D_ == 256)
    __shared__ float resid[256];
    __shared__ float red[256];
    __shared__ float quant[256];
    __shared__ float bestV[256];
    __shared__ int   bestI[256];

    float c = fmaxf(cnt[g], 1.f);
    resid[tid] = gsum[g * D_ + tid] / c;
    __syncthreads();

    for (int qs = 0; qs < Q_; ++qs) {
        // ||residual||
        red[tid] = resid[tid] * resid[tid]; __syncthreads();
        for (int s = 128; s > 0; s >>= 1) { if (tid < s) red[tid] += red[tid + s]; __syncthreads(); }
        float rnorm = fmaxf(sqrtf(red[0]), 1e-12f);
        __syncthreads();

        // argmax over cosine similarities with codebook qs
        const float* cb = codebooks + (size_t)qs * K_ * D_;
        float bv = -1e30f; int bi = 0;
        for (int code = tid; code < K_; code += 256) {
            const float* cp = cb + (size_t)code * D_;
            float dot = 0.f, nrm = 0.f;
            #pragma unroll 4
            for (int d2 = 0; d2 < D_; ++d2) { float cv = cp[d2]; dot += resid[d2] * cv; nrm += cv * cv; }
            float cs = dot / (rnorm * fmaxf(sqrtf(nrm), 1e-12f));
            if (cs > bv) { bv = cs; bi = code; }
        }
        bestV[tid] = bv; bestI[tid] = bi; __syncthreads();
        for (int s = 128; s > 0; s >>= 1) {
            if (tid < s && bestV[tid + s] > bestV[tid]) { bestV[tid] = bestV[tid + s]; bestI[tid] = bestI[tid + s]; }
            __syncthreads();
        }
        int ind = bestI[0];
        float cosSel = bestV[0];
        __syncthreads();

        // normalized selected code
        const float* cp = cb + (size_t)ind * D_;
        red[tid] = cp[tid] * cp[tid]; __syncthreads();
        for (int s = 128; s > 0; s >>= 1) { if (tid < s) red[tid] += red[tid + s]; __syncthreads(); }
        float cn = fmaxf(sqrtf(red[0]), 1e-12f);
        __syncthreads();
        quant[tid] = cp[tid] / cn;
        tokens[((size_t)g * Q_ + qs) * D_ + tid] = quant[tid];   // STE forward value
        if (tid == 0) {
            out_idx[g * Q_ + qs] = (float)ind;
            float cc = fminf(fmaxf(cosSel, -1.f), 1.f);
            atomicAdd(loss, 0.25f * (1.f - cc) / (float)NG_);
        }
        __syncthreads();

        // subtract projection
        red[tid] = resid[tid] * quant[tid]; __syncthreads();
        for (int s = 128; s > 0; s >>= 1) { if (tid < s) red[tid] += red[tid + s]; __syncthreads(); }
        float aproj = red[0];
        __syncthreads();
        resid[tid] -= aproj * quant[tid];
        __syncthreads();
    }

    // + (residual**2).mean()
    red[tid] = resid[tid] * resid[tid]; __syncthreads();
    for (int s = 128; s > 0; s >>= 1) { if (tid < s) red[tid] += red[tid + s]; __syncthreads(); }
    if (tid == 0) atomicAdd(loss, red[0] / (float)(NG_ * D_));
}

// ---------------------------------------------------------------------------
// LayerNorm over FD_ per row, h = proj + skip
// ---------------------------------------------------------------------------
__global__ __launch_bounds__(256) void layernorm_kernel(
    const float* __restrict__ h, const float* __restrict__ h2,
    const float* __restrict__ gamma, const float* __restrict__ beta,
    float* __restrict__ out)
{
    int row = blockIdx.x;
    int tid = threadIdx.x;
    __shared__ float red[256];
    const float* hp  = h  + (size_t)row * FD_;
    const float* hp2 = h2 + (size_t)row * FD_;

    float s = 0.f, ss = 0.f;
    for (int d = tid; d < FD_; d += 256) { float v = hp[d] + hp2[d]; s += v; ss += v * v; }
    red[tid] = s; __syncthreads();
    for (int k = 128; k > 0; k >>= 1) { if (tid < k) red[tid] += red[tid + k]; __syncthreads(); }
    float mean = red[0] / (float)FD_;
    __syncthreads();
    red[tid] = ss; __syncthreads();
    for (int k = 128; k > 0; k >>= 1) { if (tid < k) red[tid] += red[tid + k]; __syncthreads(); }
    float var = red[0] / (float)FD_ - mean * mean;
    __syncthreads();
    float inv = rsqrtf(var + EPS_);
    for (int d = tid; d < FD_; d += 256) {
        float v = hp[d] + hp2[d];
        out[(size_t)row * FD_ + d] = (v - mean) * inv * gamma[d] + beta[d];
    }
}

// ---------------------------------------------------------------------------
// host orchestration
// ---------------------------------------------------------------------------
static inline long cdivl(long a, long b) { return (a + b - 1) / b; }

extern "C" void kernel_launch(void* const* d_in, const int* in_sizes, int n_in,
                              void* d_out, int out_size, void* d_ws, size_t ws_size,
                              hipStream_t stream)
{
    const float* x         = (const float*)d_in[0];
    const float* edge_attr = (const float*)d_in[1];
    const int*   edge_idx  = (const int*)d_in[2];
    const int*   batch     = (const int*)d_in[3];
    const float* Wq  = (const float*)d_in[4];
    const float* bq  = (const float*)d_in[5];
    const float* Wk  = (const float*)d_in[6];
    const float* bk  = (const float*)d_in[7];
    const float* Wv  = (const float*)d_in[8];
    const float* bv  = (const float*)d_in[9];
    const float* We  = (const float*)d_in[10];
    const float* Ws  = (const float*)d_in[11];
    const float* bs  = (const float*)d_in[12];
    const float* gnw = (const float*)d_in[13];
    const float* gnb = (const float*)d_in[14];
    const float* gnm = (const float*)d_in[15];
    const float* cbs = (const float*)d_in[16];
    const float* W1  = (const float*)d_in[17];
    const float* b1  = (const float*)d_in[18];
    const float* W2  = (const float*)d_in[19];
    const float* b2  = (const float*)d_in[20];
    const float* Wsf = (const float*)d_in[21];
    const float* bsf = (const float*)d_in[22];
    const float* lng = (const float*)d_in[23];
    const float* lnb = (const float*)d_in[24];

    const int* src = edge_idx;         // row 0 of [2,E]
    const int* dst = edge_idx + E_;    // row 1

    // workspace partition
    char* ws = (char*)d_ws;
    size_t off = 0;
    auto alloc = [&](size_t bytes) -> void* {
        off = (off + 255) & ~(size_t)255;
        void* p = ws + off;
        off += bytes;
        return p;
    };
    float*    qb    = (float*)alloc((size_t)N_ * D_ * 4);
    float*    kb    = (float*)alloc((size_t)N_ * D_ * 4);
    float*    vb    = (float*)alloc((size_t)N_ * D_ * 4);
    float*    xo    = (float*)alloc((size_t)N_ * D_ * 4);
    float*    ybuf  = (float*)alloc((size_t)N_ * D_ * 4);
    float*    xA    = (float*)alloc((size_t)N_ * D_ * 4);
    float*    xB    = (float*)alloc((size_t)N_ * D_ * 4);
    float*    eb    = (float*)alloc((size_t)E_ * D_ * 4);
    float*    att   = (float*)alloc((size_t)E_ * H_ * 4);
    unsigned* mkeys = (unsigned*)alloc((size_t)N_ * H_ * 4);
    float*    denom = (float*)alloc((size_t)N_ * H_ * 4);
    float*    gsum  = (float*)alloc((size_t)NG_ * D_ * 4);
    float*    vsum  = (float*)alloc((size_t)NG_ * D_ * 4);
    float*    cnt   = (float*)alloc((size_t)NG_ * 4);
    float*    toks  = (float*)alloc((size_t)NG_ * Q_ * D_ * 4);
    float*    hid   = (float*)alloc((size_t)NG_ * Q_ * 4 * D_ * 4);
    float*    hb    = (float*)alloc((size_t)NG_ * Q_ * FD_ * 4);
    float*    h2b   = (float*)alloc((size_t)NG_ * Q_ * FD_ * 4);

    auto gemm = [&](const float* A, const float* B, const float* bias, float* Cm,
                    int M, int K, int Nc, int relu) {
        dim3 grid((unsigned)cdivl(M, BM), (unsigned)(Nc / BN));
        gemm_bf16_wmma<<<grid, 256, 0, stream>>>(A, B, bias, Cm, M, K, Nc, relu);
    };
    auto fillf = [&](float* p, float v, long n) {
        fill_f32<<<(unsigned)cdivl(n, 256), 256, 0, stream>>>(p, v, n);
    };
    auto fillu = [&](unsigned* p, unsigned v, long n) {
        fill_u32<<<(unsigned)cdivl(n, 256), 256, 0, stream>>>(p, v, n);
    };

    // per-graph node counts (constant across layers)
    fillf(cnt, 0.f, NG_);
    count_kernel<<<(unsigned)cdivl(N_, 256), 256, 0, stream>>>(batch, cnt);

    const long EH = (long)E_ * H_;
    const long ED = (long)E_ * D_;
    const long ND = (long)N_ * D_;

    for (int i = 0; i < L_; ++i) {
        const float* xin  = (i == 0) ? x  : ((i == 1) ? xA : xB);
        float*       xout = (i == 0) ? xA : ((i == 1) ? xB : xA);

        // projections (WMMA GEMMs)
        gemm(xin, Wq + (size_t)i * D_ * D_, bq + (size_t)i * D_, qb, N_, D_, D_, 0);
        gemm(xin, Wk + (size_t)i * D_ * D_, bk + (size_t)i * D_, kb, N_, D_, D_, 0);
        gemm(xin, Wv + (size_t)i * D_ * D_, bv + (size_t)i * D_, vb, N_, D_, D_, 0);
        gemm(xin, Ws + (size_t)i * D_ * D_, bs + (size_t)i * D_, xo, N_, D_, D_, 0);  // root skip
        gemm(edge_attr, We + (size_t)i * De_ * D_, nullptr, eb, E_, De_, D_, 0);

        // edge attention
        fillu(mkeys, 0u, (long)N_ * H_);
        fillf(denom, 0.f, (long)N_ * H_);
        edge_logits<<<(unsigned)cdivl(EH, 256), 256, 0, stream>>>(src, dst, qb, kb, eb, att, mkeys);
        edge_softmax_num<<<(unsigned)cdivl(EH, 256), 256, 0, stream>>>(dst, att, mkeys, denom);
        edge_message<<<(unsigned)cdivl(ED, 256), 256, 0, stream>>>(src, dst, vb, eb, att, denom, xo);

        // GraphNorm(conv_out + x_in)
        fillf(gsum, 0.f, (long)NG_ * D_);
        fillf(vsum, 0.f, (long)NG_ * D_);
        gn_pass1<<<(unsigned)cdivl(ND, 256), 256, 0, stream>>>(xo, xin, batch, ybuf, gsum);
        gn_pass2<<<(unsigned)cdivl(ND, 256), 256, 0, stream>>>(ybuf, gsum, cnt, gnm + (size_t)i * D_,
                                                               batch, xo, vsum);
        gn_pass3<<<(unsigned)cdivl(ND, 256), 256, 0, stream>>>(xo, vsum, cnt, gnw + (size_t)i * D_,
                                                               gnb + (size_t)i * D_, batch, xout);
    }
    const float* xfin = xA;  // layer 2 wrote xA

    // global mean pool
    fillf(gsum, 0.f, (long)NG_ * D_);
    pool_accum<<<(unsigned)cdivl(ND, 256), 256, 0, stream>>>(xfin, batch, gsum);

    // output layout: [NG*Q*FD] out | [NG*Q] indices | [1] loss
    float* out_f   = (float*)d_out;
    float* out_idx = out_f + (size_t)NG_ * Q_ * FD_;
    float* out_ls  = out_idx + (size_t)NG_ * Q_;
    fillf(out_ls, 0.f, 1);

    vq_kernel<<<NG_, 256, 0, stream>>>(gsum, cnt, cbs, toks, out_idx, out_ls);

    // head: relu(tokens@W1+b1)@W2+b2 + tokens@Wsf+bsf, then LayerNorm
    const int M2 = NG_ * Q_;   // 192 rows
    gemm(toks, W1,  b1,  hid, M2, D_,     4 * D_, 1);
    gemm(hid,  W2,  b2,  hb,  M2, 4 * D_, FD_,    0);
    gemm(toks, Wsf, bsf, h2b, M2, D_,     FD_,    0);
    layernorm_kernel<<<M2, 256, 0, stream>>>(hb, h2b, lng, lnb, out_f);
}
